// MultiHeadAttention_43233140802101
// MI455X (gfx1250) — matrix operand order repaired and checked by reading, compiled, not run on hardware
//
#include <hip/hip_runtime.h>
#include <hip/hip_bf16.h>

// ---------------------------------------------------------------------------
// MultiHeadAttention (shared-weight heads) for MI455X / gfx1250, wave32 WMMA.
//   B=8, T=2048, C=1024, D=64.  Folded projection: tile(O,16)@Wp == O@WpFold.
// ---------------------------------------------------------------------------

typedef __attribute__((ext_vector_type(16))) _Float16 v16h;
typedef __attribute__((ext_vector_type(8)))  float    v8f;

union F16x16 { v16h v; unsigned u[8]; _Float16 h[16]; };

__device__ __forceinline__ v8f wmma16(v16h a, v16h b, v8f c) {
  // D = A(16x32,f16) * B(32x16,f16) + C(16x16,f32)
  return __builtin_amdgcn_wmma_f32_16x16x32_f16(false, a, false, b,
                                                (short)0, c, false, false);
}

#define NB   8
#define TT   2048
#define CEMB 1024
#define HD   64
#define BT   (NB * TT)          // 16384 rows

// ---------------------------------------------------------------------------
// Kernel 1: weight prep.
//   WqT/WkT/WvT[n][k] = W[k][n]  (f16, n-major -> contiguous K-pairs for B frags)
//   WpFt[e][d] = sum_h Wp[h*64+d][e]  (folded + transposed, f16)
// ---------------------------------------------------------------------------
__global__ __launch_bounds__(256) void mha_prep(
    const float* __restrict__ Wq, const float* __restrict__ Wk,
    const float* __restrict__ Wv, const float* __restrict__ Wp,
    _Float16* __restrict__ WqT, _Float16* __restrict__ WkT,
    _Float16* __restrict__ WvT, _Float16* __restrict__ WpFt) {
  int id = blockIdx.x * 256 + threadIdx.x;     // 0 .. 65535
  if (id >= HD * CEMB) return;
  int n = id >> 10, k = id & 1023;             // transpose view
  WqT[id] = (_Float16)Wq[k * HD + n];
  WkT[id] = (_Float16)Wk[k * HD + n];
  WvT[id] = (_Float16)Wv[k * HD + n];
  int e = id >> 6, d = id & 63;                // folded-Wp view
  float s = 0.f;
#pragma unroll
  for (int h = 0; h < 16; ++h) s += Wp[(size_t)(h * HD + d) * CEMB + e];
  WpFt[(size_t)e * HD + d] = (_Float16)s;
}

// ---------------------------------------------------------------------------
// Kernel 2: Q,K,V = x @ W  (one wave per 16-token tile; K=1024 in 32-chunks).
//   Qh scaled by 1/32 (= C^-0.5).  V stored transposed Vt[b][d][t].
// ---------------------------------------------------------------------------
__global__ __launch_bounds__(256) void mha_qkv(
    const float* __restrict__ x,
    const _Float16* __restrict__ WqT, const _Float16* __restrict__ WkT,
    const _Float16* __restrict__ WvT,
    _Float16* __restrict__ Qh, _Float16* __restrict__ Kh,
    _Float16* __restrict__ Vt) {
  const int lane = threadIdx.x & 31;
  const int wave = threadIdx.x >> 5;
  const int tile = blockIdx.x * 8 + wave;      // 0..1023 (BT/16)
  const int hf   = lane >> 4;                  // lane-half
  const int l15  = lane & 15;

  v8f aq0 = {}, aq1 = {}, aq2 = {}, aq3 = {};
  v8f ak0 = {}, ak1 = {}, ak2 = {}, ak3 = {};
  v8f av0 = {}, av1 = {}, av2 = {}, av3 = {};
  v8f accq[4] = {aq0, aq1, aq2, aq3};
  v8f acck[4] = {ak0, ak1, ak2, ak3};
  v8f accv[4] = {av0, av1, av2, av3};

  const float* xrow = x + (size_t)(tile * 16 + l15) * CEMB;

  for (int kc = 0; kc < CEMB / 32; ++kc) {
    // A fragment: x rows, fp32 -> f16 in registers (contiguous k = 16*half + s order)
    F16x16 a;
#pragma unroll
    for (int i = 0; i < 8; ++i) {
      int k0 = kc * 32 + hf * 16 + 2 * i;
      float2 xv = *(const float2*)(xrow + k0);
      a.h[2 * i]     = (_Float16)xv.x;
      a.h[2 * i + 1] = (_Float16)xv.y;
    }
#pragma unroll
    for (int t = 0; t < 4; ++t) {
      int n = t * 16 + l15;
      F16x16 bq, bk, bv;
#pragma unroll
      for (int i = 0; i < 8; ++i) {
        int kk = kc * 32 + hf * 16 + 2 * i;    // B layout: K = half*16 + 2i
        bq.u[i] = *(const unsigned*)(WqT + (size_t)n * CEMB + kk);
        bk.u[i] = *(const unsigned*)(WkT + (size_t)n * CEMB + kk);
        bv.u[i] = *(const unsigned*)(WvT + (size_t)n * CEMB + kk);
      }
      accq[t] = wmma16(a.v, bq.v, accq[t]);
      acck[t] = wmma16(a.v, bk.v, acck[t]);
      accv[t] = wmma16(a.v, bv.v, accv[t]);
    }
  }

  const float qscale = 0.03125f;               // 1024^-0.5 == 1/32
  const int g0 = tile * 16;
#pragma unroll
  for (int t = 0; t < 4; ++t) {
    int n = t * 16 + l15;
#pragma unroll
    for (int r = 0; r < 8; ++r) {
      int m = r + 8 * hf;                      // C layout: M = r + 8*half
      int g = g0 + m;
      Qh[(size_t)g * HD + n] = (_Float16)(accq[t][r] * qscale);
      Kh[(size_t)g * HD + n] = (_Float16)acck[t][r];
      int b = g >> 11, pos = g & (TT - 1);
      Vt[((size_t)b * HD + n) * TT + pos] = (_Float16)accv[t][r];
    }
  }
}

// ---------------------------------------------------------------------------
// Kernel 3: causal flash attention.  One wave per 16-query tile; 32-key tiles.
// ---------------------------------------------------------------------------
__global__ __launch_bounds__(256) void mha_attn(
    const _Float16* __restrict__ Qh, const _Float16* __restrict__ Kh,
    const _Float16* __restrict__ Vt, _Float16* __restrict__ Oh) {
  __shared__ _Float16 plds[8][16 * 32];        // P tile bounce, per wave
  const int lane = threadIdx.x & 31;
  const int wave = threadIdx.x >> 5;
  const int hf   = lane >> 4;
  const int l15  = lane & 15;
  const int qtile = blockIdx.x * 8 + wave;     // 0..1023
  const int b  = qtile >> 7;                   // batch
  const int qt = qtile & 127;                  // query tile in batch

  const _Float16* Kb = Kh + (size_t)b * TT * HD;
  const _Float16* Vb = Vt + (size_t)b * HD * TT;
  _Float16* pl = &plds[wave][0];

  // Q A-fragments (d split 0..31 / 32..63), resident for the whole loop
  F16x16 aq[2];
  const _Float16* qrow = Qh + (size_t)(qtile * 16 + l15) * HD;
#pragma unroll
  for (int c = 0; c < 2; ++c)
#pragma unroll
    for (int i = 0; i < 8; ++i) {
      int k0 = c * 32 + hf * 16 + 2 * i;
      aq[c].u[i] = *(const unsigned*)(qrow + k0);
    }

  v8f acco[4] = {v8f{}, v8f{}, v8f{}, v8f{}};
  float mrow[8], lrow[8];
#pragma unroll
  for (int r = 0; r < 8; ++r) { mrow[r] = -1e30f; lrow[r] = 0.f; }

  const int nkt = ((qt * 16 + 15) >> 5) + 1;   // causal: key tiles 0..nkt-1
  for (int kt = 0; kt < nkt; ++kt) {
    const int ks = kt * 32;
    if (kt + 1 < nkt)                          // CDNA5 global_prefetch_b8
      __builtin_prefetch(Kb + (size_t)(ks + 32 + lane) * HD, 0, 3);

    // S = Q @ K^T   (two 16-col sub-tiles, contraction d=64 in 2 chunks)
    v8f s0 = {}, s1 = {};
#pragma unroll
    for (int c = 0; c < 2; ++c) {
      F16x16 b0, b1;
#pragma unroll
      for (int i = 0; i < 8; ++i) {
        int d = c * 32 + hf * 16 + 2 * i;
        b0.u[i] = *(const unsigned*)(Kb + (size_t)(ks + l15) * HD + d);
        b1.u[i] = *(const unsigned*)(Kb + (size_t)(ks + 16 + l15) * HD + d);
      }
      s0 = wmma16(aq[c].v, b0.v, s0);
      s1 = wmma16(aq[c].v, b1.v, s1);
    }

    // causal mask (only diagonal tiles need it; wave-uniform branch)
    if (ks + 31 > qt * 16) {
#pragma unroll
      for (int r = 0; r < 8; ++r) {
        int row = qt * 16 + r + 8 * hf;
        int c0  = ks + l15;
        s0[r] = (c0      > row) ? -1e30f : s0[r];
        s1[r] = (c0 + 16 > row) ? -1e30f : s1[r];
      }
    }

    // online softmax: row reductions across 16-lane halves (wave32 shuffles)
#pragma unroll
    for (int r = 0; r < 8; ++r) {
      float tm = fmaxf(s0[r], s1[r]);
#pragma unroll
      for (int off = 8; off >= 1; off >>= 1)
        tm = fmaxf(tm, __shfl_xor(tm, off, 32));
      float mnew  = fmaxf(mrow[r], tm);
      float alpha = __expf(mrow[r] - mnew);
      mrow[r] = mnew;
      float p0 = __expf(s0[r] - mnew);
      float p1 = __expf(s1[r] - mnew);
      float rs = p0 + p1;
#pragma unroll
      for (int off = 8; off >= 1; off >>= 1)
        rs += __shfl_xor(rs, off, 32);
      lrow[r] = lrow[r] * alpha + rs;
      s0[r] = p0; s1[r] = p1;
      acco[0][r] *= alpha; acco[1][r] *= alpha;
      acco[2][r] *= alpha; acco[3][r] *= alpha;
    }

    // P: C-layout -> A-layout via per-wave LDS bounce (DS ops in-order per wave)
#pragma unroll
    for (int r = 0; r < 8; ++r) {
      int m = r + 8 * hf;
      pl[m * 32 + l15]      = (_Float16)s0[r];
      pl[m * 32 + 16 + l15] = (_Float16)s1[r];
    }
    F16x16 pa;
#pragma unroll
    for (int i = 0; i < 8; ++i) {
      int col = hf * 16 + 2 * i;
      pa.u[i] = *(const unsigned*)(pl + l15 * 32 + col);
    }

    // O += P @ V   (V pre-transposed -> contiguous key pairs)
#pragma unroll
    for (int t = 0; t < 4; ++t) {
      int d = t * 16 + l15;
      F16x16 bv;
#pragma unroll
      for (int i = 0; i < 8; ++i) {
        int kk = hf * 16 + 2 * i;
        bv.u[i] = *(const unsigned*)(Vb + (size_t)d * TT + ks + kk);
      }
      acco[t] = wmma16(pa.v, bv.v, acco[t]);
    }
  }

  // normalize + store head output (f16, row-major)
#pragma unroll
  for (int r = 0; r < 8; ++r) {
    float inv = 1.0f / lrow[r];
    acco[0][r] *= inv; acco[1][r] *= inv; acco[2][r] *= inv; acco[3][r] *= inv;
  }
#pragma unroll
  for (int t = 0; t < 4; ++t)
#pragma unroll
    for (int r = 0; r < 8; ++r) {
      int m = r + 8 * hf;
      Oh[(size_t)(qtile * 16 + m) * HD + t * 16 + l15] = (_Float16)acco[t][r];
    }
}

// ---------------------------------------------------------------------------
// Kernel 4: out = O @ WpFold + bp   ([16384,64] x [64,1024], fp32 out)
// ---------------------------------------------------------------------------
__global__ __launch_bounds__(256) void mha_proj(
    const _Float16* __restrict__ Oh, const _Float16* __restrict__ WpFt,
    const float* __restrict__ bp, float* __restrict__ out) {
  const int lane = threadIdx.x & 31;
  const int wave = threadIdx.x >> 5;
  const int hf   = lane >> 4;
  const int l15  = lane & 15;
  const int wid  = blockIdx.x * 8 + wave;      // 0..16383
  const int mt   = wid >> 4;                   // row tile (16 rows)
  const int nc   = wid & 15;                   // 64-col chunk

  F16x16 ao[2];
  const _Float16* orow = Oh + (size_t)(mt * 16 + l15) * HD;
#pragma unroll
  for (int c = 0; c < 2; ++c)
#pragma unroll
    for (int i = 0; i < 8; ++i) {
      int k0 = c * 32 + hf * 16 + 2 * i;
      ao[c].u[i] = *(const unsigned*)(orow + k0);
    }

#pragma unroll
  for (int t = 0; t < 4; ++t) {
    int n = nc * 64 + t * 16 + l15;
    v8f acc = {};
#pragma unroll
    for (int c = 0; c < 2; ++c) {
      F16x16 bw;
#pragma unroll
      for (int i = 0; i < 8; ++i) {
        int d = c * 32 + hf * 16 + 2 * i;
        bw.u[i] = *(const unsigned*)(WpFt + (size_t)n * HD + d);
      }
      acc = wmma16(ao[c].v, bw.v, acc);
    }
    float bias = bp[n];
#pragma unroll
    for (int r = 0; r < 8; ++r) {
      int m = r + 8 * hf;
      out[(size_t)(mt * 16 + m) * CEMB + n] = acc[r] + bias;
    }
  }
}

// ---------------------------------------------------------------------------
extern "C" void kernel_launch(void* const* d_in, const int* in_sizes, int n_in,
                              void* d_out, int out_size, void* d_ws, size_t ws_size,
                              hipStream_t stream) {
  const float* x  = (const float*)d_in[0];
  const float* Wq = (const float*)d_in[1];
  const float* Wk = (const float*)d_in[2];
  const float* Wv = (const float*)d_in[3];
  const float* Wp = (const float*)d_in[4];
  const float* bp = (const float*)d_in[5];
  float* out = (float*)d_out;

  _Float16* ws   = (_Float16*)d_ws;
  _Float16* WqT  = ws;                          //  64*1024
  _Float16* WkT  = WqT  + HD * CEMB;
  _Float16* WvT  = WkT  + HD * CEMB;
  _Float16* WpFt = WvT  + HD * CEMB;            // 1024*64
  _Float16* Qh   = WpFt + CEMB * HD;            // 16384*64
  _Float16* Kh   = Qh   + (size_t)BT * HD;
  _Float16* Vt   = Kh   + (size_t)BT * HD;      // 8*64*2048
  _Float16* Oh   = Vt   + (size_t)BT * HD;

  mha_prep<<<(HD * CEMB + 255) / 256, 256, 0, stream>>>(Wq, Wk, Wv, Wp,
                                                        WqT, WkT, WvT, WpFt);
  mha_qkv <<<BT / 16 / 8, 256, 0, stream>>>(x, WqT, WkT, WvT, Qh, Kh, Vt);
  mha_attn<<<BT / 16 / 8, 256, 0, stream>>>(Qh, Kh, Vt, Oh);
  mha_proj<<<BT / 16 * 16 / 8, 256, 0, stream>>>(Oh, WpFt, bp, out);
}
